// SelfAttention_22436909154565
// MI455X (gfx1250) — matrix operand order repaired and checked by reading, not compiled, not run on hardware
//
#include <hip/hip_runtime.h>
#include <hip/hip_bf16.h>

// ---------------------------------------------------------------------------
// Self-attention + FFN block for MI455X (gfx1250), wave32, WMMA f16->f32.
// B=16, S=1024, H=768.  All GEMMs via v_wmma_f32_16x16x32_f16.
// ---------------------------------------------------------------------------

typedef _Float16 v16h __attribute__((ext_vector_type(16)));
typedef _Float16 v8h  __attribute__((ext_vector_type(8)));
typedef _Float16 v4h  __attribute__((ext_vector_type(4)));
typedef float    v8f  __attribute__((ext_vector_type(8)));

#define BATCH 16
#define SEQ   1024
#define HID   768

// LDS transpose tile: per-wave 64 rows (h) x 32 cols (s), row stride 40 halves
// (80 B: 16B-aligned rows, 20-bank lane stride -> conflict-free).
#define TPOSE_STRIDE 40
#define TPOSE_WAVE_HALVES (64 * TPOSE_STRIDE)

union AFrag { v16h v; v8h h[2]; };

__device__ __forceinline__ v8f wmma_f16f32(v16h a, v16h b, v8f c) {
  // 8 args: (neg_a, A, neg_b, B, c_mod, C, reuse_a, reuse_b)
  return __builtin_amdgcn_wmma_f32_16x16x32_f16(false, a, false, b, (short)0, c,
                                                false, false);
}

// ---------------------------------------------------------------------------
// f32 -> f16 conversion (vectorized x4)
// ---------------------------------------------------------------------------
__global__ void f32_to_f16_kernel(const float* __restrict__ src,
                                  _Float16* __restrict__ dst, int n4) {
  int i = blockIdx.x * blockDim.x + threadIdx.x;
  int stride = gridDim.x * blockDim.x;
  const float4* s4 = (const float4*)src;
  v4h* d4 = (v4h*)dst;
  for (; i < n4; i += stride) {
    float4 f = s4[i];
    v4h h;
    h.x = (_Float16)f.x; h.y = (_Float16)f.y;
    h.z = (_Float16)f.z; h.w = (_Float16)f.w;
    d4[i] = h;
  }
}

// ---------------------------------------------------------------------------
// Generic NT GEMM:  C[m,n] = sum_k A[m,k] * W[n,k]  (+bias, relu, residual)
// A: f16 [M x K] row-major (lda = K), W: f16 [N x K] row-major (ldw = K).
// Workgroup: 256 threads = 8 waves; block tile 128(M) x 128(N);
// wave tile 32 x 64 (2x4 WMMA 16x16 accumulators). M,N multiples of 128.
//
// Per-lane k order used here (contiguous k = 16h + s, not the ISA table order; A and B match):
//   A 16x32:  lane L(<16) row M=L holds K {0..15}; lane L+16 holds
//             K {16..31}  -> two contiguous 16B chunks per lane.
//   B 32x16:  lane L(<16) col N=L holds K {0..15}; lane L+16 holds K {16..31}
//             -> one contiguous 32B chunk per lane (NT: B col = W row).
//
// Transposed output path (oT != null): wave tile is staged in LDS (dynamic
// shared, 40 KB for the whole block) and written back as contiguous 64 B
// column runs of V^T instead of scattered 2 B stores.
// ---------------------------------------------------------------------------
__global__ void __launch_bounds__(256)
gemm_nt_f16(const _Float16* __restrict__ A, int lda, long long aBatch,
            const _Float16* __restrict__ W, int ldw, long long wBatch,
            const float* __restrict__ bias,
            float* __restrict__ o32, long long oBatch,
            _Float16* __restrict__ o16,
            _Float16* __restrict__ oT, long long oTBatch,  // per-1024-row-chunk transpose
            const float* __restrict__ resid,
            int K, int N, int relu) {
  extern __shared__ _Float16 ldsT[];  // used only when oT != nullptr

  const int z = blockIdx.z;
  A += (size_t)z * aBatch;
  W += (size_t)z * wBatch;
  if (o32)   o32   += (size_t)z * oBatch;
  if (o16)   o16   += (size_t)z * oBatch;
  if (resid) resid += (size_t)z * oBatch;

  const int tid  = threadIdx.x;
  const int lane = tid & 31;
  const int wave = tid >> 5;
  const int l15  = lane & 15;
  const int hi   = lane >> 4;

  const int mWave = blockIdx.y * 128 + (wave & 3) * 32;   // 4 waves along M
  const int nWave = blockIdx.x * 128 + (wave >> 2) * 64;  // 2 waves along N

  v8f acc[2][4] = {};

  for (int k0 = 0; k0 < K; k0 += 32) {
    AFrag a[2];
#pragma unroll
    for (int mt = 0; mt < 2; ++mt) {
      const _Float16* ar =
          A + (size_t)(mWave + mt * 16 + l15) * lda + k0 + hi * 16;
      a[mt].h[0] = *(const v8h*)ar;
      a[mt].h[1] = *(const v8h*)(ar + 8);
    }
#pragma unroll
    for (int nt = 0; nt < 4; ++nt) {
      const _Float16* br =
          W + (size_t)(nWave + nt * 16 + l15) * ldw + k0 + hi * 16;
      v16h bf = *(const v16h*)br;
#pragma unroll
      for (int mt = 0; mt < 2; ++mt)
        acc[mt][nt] = wmma_f16f32(a[mt].v, bf, acc[mt][nt]);
    }
  }

  _Float16* T = ldsT + wave * TPOSE_WAVE_HALVES;

  // Epilogue.  C/D layout: lane<16 -> (M = mTile+v, N = nTile+l15);
  // lane>=16 -> M = mTile+8+v.
#pragma unroll
  for (int mt = 0; mt < 2; ++mt) {
    const int r0 = mWave + mt * 16 + hi * 8;
#pragma unroll
    for (int nt = 0; nt < 4; ++nt) {
      const int col = nWave + nt * 16 + l15;
      const float bv = bias ? bias[col] : 0.0f;
#pragma unroll
      for (int v = 0; v < 8; ++v) {
        float x = acc[mt][nt][v] + bv;
        if (relu) x = fmaxf(x, 0.0f);
        const int row = r0 + v;
        const size_t idx = (size_t)row * N + col;
        if (resid) x += resid[idx];
        if (o32) o32[idx] = x;
        if (o16) o16[idx] = (_Float16)x;
        if (oT) {
          // stage transposed in LDS: T[h_local][s_local]
          const int hl = nt * 16 + l15;               // 0..63
          const int sl = mt * 16 + hi * 8 + v;        // 0..31
          T[hl * TPOSE_STRIDE + sl] = (_Float16)x;
        }
      }
    }
  }

  if (oT) {
    __syncthreads();  // uniform: oT is a kernel argument
    const int bc = mWave >> 10;        // 1024-row batch chunk
    const int s0 = mWave & 1023;       // s-base of this wave tile
#pragma unroll
    for (int rep = 0; rep < 2; ++rep) {
      const int hl = lane + rep * 32;  // 0..63 within wave tile
      const _Float16* src = T + hl * TPOSE_STRIDE;
      _Float16* dst =
          oT + (size_t)bc * oTBatch + (size_t)(nWave + hl) * SEQ + s0;
#pragma unroll
      for (int c = 0; c < 4; ++c)
        *(v8h*)(dst + c * 8) = *(const v8h*)(src + c * 8);
    }
  }
}

// ---------------------------------------------------------------------------
// Attention scores + masked softmax, one workgroup per (batch, 16 q-rows).
// 8 waves; wave w computes score columns [w*128, w*128+128) via WMMA,
// writes the 16 x 1024 f32 strip to LDS, then half-wave-per-row softmax.
// P[b,s,t] = m_s * m_t * softmax_t(score) stored as f16.
// ---------------------------------------------------------------------------
__global__ void __launch_bounds__(256)
attn_softmax(const _Float16* __restrict__ Qh, const _Float16* __restrict__ Kh,
             const int* __restrict__ mask, _Float16* __restrict__ Ph) {
  __shared__ float sc[16][SEQ];  // 64 KB

  const int b  = blockIdx.y;
  const int q0 = blockIdx.x * 16;
  const int tid  = threadIdx.x;
  const int lane = tid & 31;
  const int wave = tid >> 5;
  const int l15  = lane & 15;
  const int hi   = lane >> 4;

  const _Float16* Qb = Qh + ((size_t)b * SEQ + q0) * HID;
  const _Float16* Kb = Kh + (size_t)b * SEQ * HID;
  const int* mb = mask + b * SEQ;

  v8f acc[8] = {};
  for (int k0 = 0; k0 < HID; k0 += 32) {
    AFrag a;
    const _Float16* ar = Qb + (size_t)l15 * HID + k0 + hi * 16;
    a.h[0] = *(const v8h*)ar;
    a.h[1] = *(const v8h*)(ar + 8);
#pragma unroll
    for (int tt = 0; tt < 8; ++tt) {
      const int trow = wave * 128 + tt * 16 + l15;
      const _Float16* br = Kb + (size_t)trow * HID + k0 + hi * 16;
      v16h bf = *(const v16h*)br;
      acc[tt] = wmma_f16f32(a.v, bf, acc[tt]);
    }
  }

  const float scale = 0.03608439182435161f;  // 1/sqrt(768)
#pragma unroll
  for (int tt = 0; tt < 8; ++tt) {
    const int col = wave * 128 + tt * 16 + l15;
    const bool keep = mb[col] != 0;
    const int r0 = hi * 8;
#pragma unroll
    for (int v = 0; v < 8; ++v)
      sc[r0 + v][col] = keep ? acc[tt][v] * scale : -1.0e30f;
  }
  __syncthreads();

  // Softmax: row r handled by 16 consecutive threads (one half-wave).
  const int r  = tid >> 4;
  const int c0 = tid & 15;
  const float ms = (mb[q0 + r] != 0) ? 1.0f : 0.0f;

  float mx = -3.4e38f;
#pragma unroll 4
  for (int j = 0; j < 64; ++j) mx = fmaxf(mx, sc[r][c0 + 16 * j]);
#pragma unroll
  for (int d = 1; d < 16; d <<= 1) mx = fmaxf(mx, __shfl_xor(mx, d, 16));

  float sum = 0.0f;
#pragma unroll 4
  for (int j = 0; j < 64; ++j) {
    const int c = c0 + 16 * j;
    const float p = __expf(sc[r][c] - mx);
    sc[r][c] = p;
    sum += p;
  }
#pragma unroll
  for (int d = 1; d < 16; d <<= 1) sum += __shfl_xor(sum, d, 16);

  const float inv = ms / sum;  // sum >= 1 always (max term contributes 1)
  _Float16* Pr = Ph + ((size_t)b * SEQ + q0 + r) * SEQ;
#pragma unroll 4
  for (int j = 0; j < 64; ++j) {
    const int c = c0 + 16 * j;
    const float mt = (mb[c] != 0) ? 1.0f : 0.0f;
    Pr[c] = (_Float16)(sc[r][c] * inv * mt);
  }
}

// ---------------------------------------------------------------------------
// Host-side orchestration
// ---------------------------------------------------------------------------
extern "C" void kernel_launch(void* const* d_in, const int* in_sizes, int n_in,
                              void* d_out, int out_size, void* d_ws,
                              size_t ws_size, hipStream_t stream) {
  const float* x    = (const float*)d_in[0];
  const int*   mask = (const int*)d_in[1];
  const float* Wq = (const float*)d_in[2];  const float* bq = (const float*)d_in[3];
  const float* Wk = (const float*)d_in[4];  const float* bk = (const float*)d_in[5];
  const float* Wv = (const float*)d_in[6];  const float* bv = (const float*)d_in[7];
  const float* W1 = (const float*)d_in[8];  const float* b1 = (const float*)d_in[9];
  const float* W2 = (const float*)d_in[10]; const float* b2 = (const float*)d_in[11];
  float* out = (float*)d_out;

  const size_t MROWS = (size_t)BATCH * SEQ;         // 16384
  const size_t XE  = MROWS * HID;                   // 12,582,912 elems
  const size_t WE  = (size_t)HID * HID;             // 589,824 elems
  const size_t PE  = (size_t)BATCH * SEQ * SEQ;     // 16,777,216 elems

  char* ws = (char*)d_ws;
  size_t off = 0;
  _Float16* Xh  = (_Float16*)(ws + off); off += XE * 2;
  _Float16* Wqh = (_Float16*)(ws + off); off += WE * 2;
  _Float16* Wkh = (_Float16*)(ws + off); off += WE * 2;
  _Float16* Wvh = (_Float16*)(ws + off); off += WE * 2;
  _Float16* W1h = (_Float16*)(ws + off); off += WE * 2;
  _Float16* W2h = (_Float16*)(ws + off); off += WE * 2;
  _Float16* Qh  = (_Float16*)(ws + off); off += XE * 2;
  _Float16* Kh  = (_Float16*)(ws + off); off += XE * 2;
  _Float16* Vth = (_Float16*)(ws + off); off += XE * 2;   // [B][H][S] (V^T)
  _Float16* Ph  = (_Float16*)(ws + off); off += PE * 2;   // [B][S][S]
  _Float16* Ofh = (_Float16*)(ws + off); off += XE * 2;   // attn out f16
  _Float16* H1h = (_Float16*)(ws + off); off += XE * 2;   // ffn hidden f16
  (void)off; (void)ws_size;

  dim3 blk(256);
  const size_t tposeLds = 8 * TPOSE_WAVE_HALVES * sizeof(_Float16);  // 40 KB

  // --- converts ---
  f32_to_f16_kernel<<<2048, 256, 0, stream>>>(x,  Xh,  (int)(XE / 4));
  f32_to_f16_kernel<<<288,  256, 0, stream>>>(Wq, Wqh, (int)(WE / 4));
  f32_to_f16_kernel<<<288,  256, 0, stream>>>(Wk, Wkh, (int)(WE / 4));
  f32_to_f16_kernel<<<288,  256, 0, stream>>>(Wv, Wvh, (int)(WE / 4));
  f32_to_f16_kernel<<<288,  256, 0, stream>>>(W1, W1h, (int)(WE / 4));
  f32_to_f16_kernel<<<288,  256, 0, stream>>>(W2, W2h, (int)(WE / 4));

  const dim3 gProj(HID / 128, (unsigned)(MROWS / 128), 1);  // (6,128,1)

  // --- Q = relu(X Wq^T + bq) -> f16 ---
  gemm_nt_f16<<<gProj, blk, 0, stream>>>(Xh, HID, 0, Wqh, HID, 0, bq,
      nullptr, 0, Qh, nullptr, 0, nullptr, HID, HID, 1);
  // --- K = relu(X Wk^T + bk) -> f16 ---
  gemm_nt_f16<<<gProj, blk, 0, stream>>>(Xh, HID, 0, Wkh, HID, 0, bk,
      nullptr, 0, Kh, nullptr, 0, nullptr, HID, HID, 1);
  // --- V = X Wv^T + bv -> f16, stored transposed per batch: Vth[b][h][s],
  //     via LDS-staged 64 B column-run stores ---
  gemm_nt_f16<<<gProj, blk, tposeLds, stream>>>(Xh, HID, 0, Wvh, HID, 0, bv,
      nullptr, 0, nullptr, Vth, (long long)HID * SEQ, nullptr, HID, HID, 0);

  // --- P = mask-softmax(Q K^T / sqrt(H)) -> f16 ---
  attn_softmax<<<dim3(SEQ / 16, BATCH), blk, 0, stream>>>(Qh, Kh, mask, Ph);

  // --- attn out = P V  (NT against Vth): f32 -> d_out, f16 -> Ofh ---
  gemm_nt_f16<<<dim3(HID / 128, SEQ / 128, BATCH), blk, 0, stream>>>(
      Ph, SEQ, (long long)SEQ * SEQ, Vth, SEQ, (long long)HID * SEQ, nullptr,
      out, (long long)SEQ * HID, Ofh, nullptr, 0, nullptr, SEQ, HID, 0);

  // --- h1 = relu(attnOut W1^T + b1) -> f16 ---
  gemm_nt_f16<<<gProj, blk, 0, stream>>>(Ofh, HID, 0, W1h, HID, 0, b1,
      nullptr, 0, H1h, nullptr, 0, nullptr, HID, HID, 1);
  // --- out = h1 W2^T + b2 + attnOut(residual, in-place in d_out) ---
  gemm_nt_f16<<<gProj, blk, 0, stream>>>(H1h, HID, 0, W2h, HID, 0, b2,
      out, 0, nullptr, nullptr, 0, out, HID, HID, 0);
}
